// NVGPTCoreAttention_49486613184797
// MI455X (gfx1250) — matrix operand order repaired and checked by reading, not compiled, not run on hardware
//
#include <hip/hip_runtime.h>
#include <math.h>

// CDNA5 / gfx1250 flash-attention (causal), wave32, WMMA f16->f32.
// Transposed formulation: S^T = K*Q^T, ctx^T = V^T * P^T; each lane owns one
// query column -> in-register softmax, no LDS, one shfl_xor(16) pair swap.
// Loop split into mask-free steady state + exactly one masked tail tile.
// SQ=2048, B=1, H=16, D=64, HIDDEN=1024, fp32 in/out.

typedef __attribute__((ext_vector_type(16))) _Float16 v16h;
typedef __attribute__((ext_vector_type(8)))  _Float16 v8h;
typedef __attribute__((ext_vector_type(4)))  _Float16 v4h;
typedef __attribute__((ext_vector_type(8)))  float    v8f;
typedef __attribute__((ext_vector_type(8)))  int      v8i;

#define SQ_     2048
#define H_      16
#define D_      64
#define HIDDEN_ (H_ * D_)

// ---------------- pre-pass 1: K f32 -> f16 (same layout [sq][h][d]) ----------------
__global__ __launch_bounds__(256)
void cvt_k_kernel(const float* __restrict__ X, _Float16* __restrict__ Y) {
    const int i = (blockIdx.x * 256 + threadIdx.x) * 4;   // n = SQ*HIDDEN, exact
    const float4 v = *(const float4*)(X + i);
    v4h h; h[0] = (_Float16)v.x; h[1] = (_Float16)v.y;
           h[2] = (_Float16)v.z; h[3] = (_Float16)v.w;
    *(v4h*)(Y + i) = h;
}

// ---------------- pre-pass 2: V f32 [sq][h][d] -> f16 transposed [h][d][sq] --------
__global__ __launch_bounds__(256)
void cvt_vt_kernel(const float* __restrict__ V, _Float16* __restrict__ Vt) {
    const int t = blockIdx.x * 256 + threadIdx.x;   // SQ*H*(D/4) threads
    const int s   = t >> 8;
    const int rem = t & 255;
    const int h   = rem >> 4;
    const int d0  = (rem & 15) * 4;
    const float4 v = *(const float4*)(V + (size_t)s * HIDDEN_ + h * D_ + d0);
    _Float16* base = Vt + ((size_t)h * D_ + d0) * SQ_ + s;
    base[0 * SQ_] = (_Float16)v.x;
    base[1 * SQ_] = (_Float16)v.y;
    base[2 * SQ_] = (_Float16)v.z;
    base[3 * SQ_] = (_Float16)v.w;
}

__device__ __forceinline__ unsigned pack2h(float x, float y) {
    const unsigned short ux = __builtin_bit_cast(unsigned short, (_Float16)x);
    const unsigned short uy = __builtin_bit_cast(unsigned short, (_Float16)y);
    return (unsigned)ux | ((unsigned)uy << 16);
}

// ---------------- one 32-key KV tile ----------------
template <bool MASKED>
__device__ __forceinline__ void kv_tile(const _Float16* __restrict__ Kbase,
                                        const _Float16* __restrict__ Vbase,
                                        const v16h& qb0, const v16h& qb1,
                                        int kb, int q, int half, int l16,
                                        v8f& acc0, v8f& acc1, v8f& acc2, v8f& acc3,
                                        float& m_run, float& l_run)
{
    // ---- S^T = K * Q^T : two C-frags (keys kb..+15 and kb+16..+31) ----
    // A 16x32: M=key=l16(+g*16), element i -> K(d) = half*8 + (i&7) + (i>>3)*16
    v8f st0 = {}, st1 = {};
#pragma unroll
    for (int g = 0; g < 2; ++g) {
        const _Float16* krow = Kbase + (size_t)(kb + g * 16 + l16) * HIDDEN_;
        const v8h a0 = *(const v8h*)(krow + half * 8);
        const v8h a1 = *(const v8h*)(krow + half * 8 + 16);
        const v8h a2 = *(const v8h*)(krow + 32 + half * 8);
        const v8h a3 = *(const v8h*)(krow + 32 + half * 8 + 16);
        v16h ka, kc;
#pragma unroll
        for (int j = 0; j < 8; ++j) {
            ka[j] = a0[j]; ka[8 + j] = a1[j];
            kc[j] = a2[j]; kc[8 + j] = a3[j];
        }
        v8f s = {};
        s = __builtin_amdgcn_wmma_f32_16x16x32_f16(false, ka, false, qb0,
                                                   (short)0, s, false, false);
        s = __builtin_amdgcn_wmma_f32_16x16x32_f16(false, kc, false, qb1,
                                                   (short)0, s, false, false);
        if (g == 0) st0 = s; else st1 = s;
    }

    // ---- (optional) causal mask + online softmax (log2 domain) ----
    // st0 row r -> key kb+half*8+r ; st1 row r -> key kb+16+half*8+r.
    float sc[16];
#pragma unroll
    for (int r = 0; r < 8; ++r) {
        if (MASKED) {
            const int key0 = kb + half * 8 + r;
            const int key1 = kb + 16 + half * 8 + r;
            sc[r]     = (key0 <= q) ? st0[r] : -1.0e5f;
            sc[8 + r] = (key1 <= q) ? st1[r] : -1.0e5f;
        } else {
            sc[r]     = st0[r];
            sc[8 + r] = st1[r];
        }
    }
    float mx = sc[0];
#pragma unroll
    for (int i = 1; i < 16; ++i) mx = fmaxf(mx, sc[i]);
    mx = fmaxf(mx, __shfl_xor(mx, 16, 32));          // combine with partner lane
    const float mnew = fmaxf(mx, m_run);
    float p[16], rs = 0.0f;
#pragma unroll
    for (int i = 0; i < 16; ++i) { p[i] = exp2f(sc[i] - mnew); rs += p[i]; }
    rs += __shfl_xor(rs, 16, 32);
    const float alpha = exp2f(m_run - mnew);
    l_run = l_run * alpha + rs;
    m_run = mnew;
#pragma unroll
    for (int r = 0; r < 8; ++r) {
        acc0[r] *= alpha; acc1[r] *= alpha;
        acc2[r] *= alpha; acc3[r] *= alpha;
    }

    // ---- assemble P^T B-fragment (keys x queries) via partner swap ----
    unsigned a[4], b[4], oa[4], ob[4];
#pragma unroll
    for (int j = 0; j < 4; ++j) {
        a[j] = pack2h(p[2 * j],     p[2 * j + 1]);       // keys kb+half*8+..
        b[j] = pack2h(p[8 + 2 * j], p[8 + 2 * j + 1]);   // keys kb+16+half*8+..
    }
#pragma unroll
    for (int j = 0; j < 4; ++j) {
        oa[j] = __shfl_xor(a[j], 16, 32);
        ob[j] = __shfl_xor(b[j], 16, 32);
    }
    v8i iv;
#pragma unroll
    for (int j = 0; j < 4; ++j) {
        // half==0: keys 0..7 own(a), keys 8..15 partner(oa)
        // half==1: keys 16..23 partner(ob), keys 24..31 own(b)
        iv[j]     = (int)(half ? ob[j] : a[j]);
        iv[4 + j] = (int)(half ? b[j]  : oa[j]);
    }
    const v16h pb = __builtin_bit_cast(v16h, iv);

    // ---- ctx^T += V^T * P^T : four d-chunks ----
    // A 16x32: M=d=c*16+l16, element i -> key kb + half*16 + i
#pragma unroll
    for (int c = 0; c < 4; ++c) {
        const _Float16* vrow = Vbase + (size_t)(c * 16 + l16) * SQ_ + kb + half * 16;
        const v8h w0 = *(const v8h*)(vrow);
        const v8h w1 = *(const v8h*)(vrow + 8);
        v16h va;
#pragma unroll
        for (int j = 0; j < 8; ++j) { va[j] = w0[j]; va[8 + j] = w1[j]; }
        if      (c == 0) acc0 = __builtin_amdgcn_wmma_f32_16x16x32_f16(false, va, false, pb, (short)0, acc0, false, false);
        else if (c == 1) acc1 = __builtin_amdgcn_wmma_f32_16x16x32_f16(false, va, false, pb, (short)0, acc1, false, false);
        else if (c == 2) acc2 = __builtin_amdgcn_wmma_f32_16x16x32_f16(false, va, false, pb, (short)0, acc2, false, false);
        else             acc3 = __builtin_amdgcn_wmma_f32_16x16x32_f16(false, va, false, pb, (short)0, acc3, false, false);
    }
}

// ---------------- main flash-attention kernel (no LDS) ----------------
__global__ __launch_bounds__(128)
void fa_causal_wmma_kernel(const float* __restrict__ Q,
                           const _Float16* __restrict__ Kh,   // [sq][h][d] f16
                           const _Float16* __restrict__ Vt,   // [h][d][sq] f16
                           float* __restrict__ O)
{
    const int lane  = threadIdx.x & 31;
    const int wib   = threadIdx.x >> 5;
    const int w     = blockIdx.x * 4 + wib;
    const int qtile = w & (SQ_ / 16 - 1);
    const int head  = w >> 7;
    const int q0    = qtile * 16;

    const int half = lane >> 4;
    const int l16  = lane & 15;
    const int q    = q0 + l16;

    // ---- Q B-fragments (B 32x16: N=query=l16, element i -> K row d = half*8 + (i&7) + (i>>3)*16) ----
    // Fold 1/sqrt(64) * log2(e) into Q so scores land in the log2 domain pre-scaled.
    const float qscale = 0.125f * 1.44269504f;
    v16h qb0, qb1;   // d chunks [0,32) and [32,64)
    {
        const float* qrow = Q + (size_t)q * HIDDEN_ + head * D_ + half * 8;
#pragma unroll
        for (int i = 0; i < 16; ++i) {
            qb0[i] = (_Float16)(qrow[(i & 7) + (i >> 3) * 16]      * qscale);
            qb1[i] = (_Float16)(qrow[32 + (i & 7) + (i >> 3) * 16] * qscale);
        }
    }

    v8f acc0 = {}, acc1 = {}, acc2 = {}, acc3 = {};
    float m_run = -INFINITY, l_run = 0.0f;

    const _Float16* Kbase = Kh + head * D_;
    const _Float16* Vbase = Vt + (size_t)head * D_ * SQ_;

    // Fully-unmasked tiles: kb+31 <= q0  ->  nfull = (q0+1)>>5.
    // Total tiles (q0+47)>>5 == nfull+1 always: exactly one masked tail tile.
    const int nfull = (q0 + 1) >> 5;
    for (int t = 0; t < nfull; ++t)
        kv_tile<false>(Kbase, Vbase, qb0, qb1, t * 32, q, half, l16,
                       acc0, acc1, acc2, acc3, m_run, l_run);
    kv_tile<true>(Kbase, Vbase, qb0, qb1, nfull * 32, q, half, l16,
                  acc0, acc1, acc2, acc3, m_run, l_run);

    // ---- write out: lane owns query column q; frag c row r -> d = c*16+half*8+r ----
    const float inv = 1.0f / l_run;
    float* orow = O + (size_t)q * HIDDEN_ + head * D_;
#pragma unroll
    for (int c = 0; c < 4; ++c) {
        const v8f& a = (c == 0) ? acc0 : (c == 1) ? acc1 : (c == 2) ? acc2 : acc3;
        float4 lo, hi;
        lo.x = a[0] * inv; lo.y = a[1] * inv; lo.z = a[2] * inv; lo.w = a[3] * inv;
        hi.x = a[4] * inv; hi.y = a[5] * inv; hi.z = a[6] * inv; hi.w = a[7] * inv;
        float* dst = orow + c * 16 + half * 8;
        *(float4*)(dst)     = lo;
        *(float4*)(dst + 4) = hi;
    }
}

extern "C" void kernel_launch(void* const* d_in, const int* in_sizes, int n_in,
                              void* d_out, int out_size, void* d_ws, size_t ws_size,
                              hipStream_t stream) {
    (void)in_sizes; (void)n_in; (void)out_size; (void)ws_size;
    const float* Q = (const float*)d_in[0];
    const float* K = (const float*)d_in[1];
    const float* V = (const float*)d_in[2];
    float* O = (float*)d_out;

    _Float16* Kh = (_Float16*)d_ws;                       // 4 MB
    _Float16* Vt = Kh + (size_t)SQ_ * HIDDEN_;            // 4 MB, layout [h][d][sq]

    const int n4 = (SQ_ * HIDDEN_) / 4;                   // 524288
    cvt_k_kernel <<<n4 / 256, 256, 0, stream>>>(K, Kh);
    cvt_vt_kernel<<<n4 / 256, 256, 0, stream>>>(V, Vt);

    const int blocks = (H_ * (SQ_ / 16)) / 4;             // 512
    fa_causal_wmma_kernel<<<blocks, 128, 0, stream>>>(Q, Kh, Vt, O);
}
